// SMC_Transformer_16655883174780
// MI455X (gfx1250) — matrix operand order repaired and checked by reading, not compiled, not run on hardware
//
#include <hip/hip_runtime.h>
#include <hip/hip_bf16.h>

#define DEV __device__ __forceinline__

typedef __attribute__((ext_vector_type(16))) _Float16 v16h;
typedef __attribute__((ext_vector_type(8)))  float    v8f;

constexpr int Bb = 8, Pp = 10, Ss = 160, FXx = 11, Dd = 64, DFFf = 256;
constexpr int NBP  = Bb * Pp;     // 80 sequences
constexpr int ROWS = NBP * Ss;    // 12800 token rows
constexpr int MT   = Ss / 16;     // 10 row tiles per sequence

DEV int lane_id() { return (int)(threadIdx.x & 31); }

// ---- CDNA5 wave32 WMMA fragment packers (contiguous k = 16h+e order, same for A and B) ----

// A 16x32 f16: lanes 0-15 hold M=0..15; element e -> K=16h+e
// (one contiguous 32B run per lane -> compiler merges into b128 loads)
DEV v16h a_frag(const _Float16* base, int ld, int kbase) {
  int l = lane_id(), h = l >> 4, m = l & 15;
  v16h a;
#pragma unroll
  for (int e = 0; e < 16; ++e) {
    int k = kbase + h * 16 + e;
    a[e] = base[m * ld + k];
  }
  return a;
}

// B 32x16 f16 where B[k][n] = src[n*ld + k] (transposed source; contiguous in e)
DEV v16h bT_frag_h(const _Float16* src, int ld, int kbase) {
  int l = lane_id(), h = l >> 4, n = l & 15;
  v16h b;
#pragma unroll
  for (int e = 0; e < 16; ++e) {
    int k = kbase + h * 16 + e;
    b[e] = src[n * ld + k];
  }
  return b;
}

// B fragment from pre-packed weight: one aligned 32-byte load per lane
DEV v16h b_frag_packed(const _Float16* P, int kc, int nt, int NT) {
  return *(const v16h*)(P + ((((kc * NT + nt) * 32) + lane_id()) << 4));
}

DEV v8f wmma32(const v16h& a, const v16h& b, const v8f& c) {
  return __builtin_amdgcn_wmma_f32_16x16x32_f16(false, a, false, b, (short)0, c,
                                                false, false);
}

// ------------- Kernel 0: pack f32 weight (Kdim x Ndim) into f16 fragments -------------
// Layout: P[((kc*NT + nt)*32 + lane)*16 + e] = W[kc*32 + 16*(lane>>4) + e][nt*16 + (lane&15)]
__global__ void smc_packw_kernel(const float* __restrict__ W, _Float16* __restrict__ P,
                                 int Kdim, int Ndim) {
  int idx = blockIdx.x * blockDim.x + threadIdx.x;
  if (idx >= Kdim * Ndim) return;
  int e = idx & 15, lane = (idx >> 4) & 31, frag = idx >> 9;
  int NT = Ndim >> 4;
  int kc = frag / NT, nt = frag - kc * NT;
  int h = lane >> 4, n = lane & 15;
  int k = kc * 32 + h * 16 + e;
  P[idx] = (_Float16)W[k * Ndim + nt * 16 + n];
}

// ---------------- Kernel 1: x = (in @ Wproj + b) * sqrt(D) ----------------
__global__ void smc_proj_kernel(const float* __restrict__ inp,
                                const float* __restrict__ Wp,
                                const float* __restrict__ bp,
                                float* __restrict__ x32,
                                _Float16* __restrict__ x16) {
  int idx = blockIdx.x * blockDim.x + threadIdx.x;
  if (idx >= ROWS * Dd) return;
  int row = idx >> 6, d = idx & 63;
  const float* ir = inp + row * FXx;
  float acc = bp[d];
#pragma unroll
  for (int f = 0; f < FXx; ++f) acc += ir[f] * Wp[f * Dd + d];
  acc *= 8.0f;  // sqrt(64)
  x32[idx] = acc;
  x16[idx] = (_Float16)acc;
}

// ---------------- Kernel 2: Q,K,V projections (WMMA, M=16 tiles) ----------------
// Compile-time output selection: no runtime null checks / branches in epilogue.
template <bool HAS_H, bool HAS_F, bool HAS_HT>
DEV void gemm_w64(const v16h& a0, const v16h& a1, const _Float16* Wp,
                  const float* bias, float h_scale,
                  _Float16* out_h, float* out_f, _Float16* out_ht, int ht_ld) {
  int l = lane_id(), h = l >> 4, n = l & 15;
#pragma unroll
  for (int nt = 0; nt < 4; ++nt) {
    v8f c = {};
    c = wmma32(a0, b_frag_packed(Wp, 0, nt, 4), c);
    c = wmma32(a1, b_frag_packed(Wp, 1, nt, 4), c);
    int ng = nt * 16 + n;
    float bv = bias[ng];
#pragma unroll
    for (int f = 0; f < 8; ++f) {
      int m = f + 8 * h;
      float v = c[f] + bv;
      if constexpr (HAS_H)  out_h[m * Dd + ng] = (_Float16)(v * h_scale);
      if constexpr (HAS_F)  out_f[m * Dd + ng] = v;
      if constexpr (HAS_HT) out_ht[ng * ht_ld + m] = (_Float16)v;  // transposed copy
    }
  }
}

__global__ void __launch_bounds__(32) smc_qkv_kernel(
    const _Float16* __restrict__ x16,
    const _Float16* __restrict__ Wqp, const float* __restrict__ bq,
    const _Float16* __restrict__ Wkp, const float* __restrict__ bk,
    const _Float16* __restrict__ Wvp, const float* __restrict__ bv,
    _Float16* __restrict__ Qh, _Float16* __restrict__ Kh, _Float16* __restrict__ VhT,
    float* __restrict__ Kout, float* __restrict__ Vout) {
  int tile = blockIdx.x;                       // 0..799
  int bp = tile / MT, s0 = (tile - bp * MT) * 16;
  const _Float16* A = x16 + tile * 16 * Dd;
  v16h a0 = a_frag(A, Dd, 0), a1 = a_frag(A, Dd, 32);
  // Q: fold softmax scale 1/sqrt(D) into the f16 copy
  gemm_w64<true, false, false>(a0, a1, Wqp, bq, 0.125f,
                               Qh + tile * 1024, nullptr, nullptr, 0);
  gemm_w64<true, true, false>(a0, a1, Wkp, bk, 1.0f,
                              Kh + tile * 1024, Kout + tile * 1024, nullptr, 0);
  // V: f32 output + TRANSPOSED f16 copy VhT[d][s] (contiguous attn@V B-frags,
  //    and the m-contiguous stores merge into wide stores)
  gemm_w64<false, true, true>(a0, a1, Wvp, bv, 1.0f, nullptr, Vout + tile * 1024,
                              VhT + bp * Dd * Ss + s0, Ss);
}

// ---------------- Kernel 3: causal logits = Q K^T (WMMA) ----------------
__global__ void __launch_bounds__(32) smc_scores_kernel(
    const _Float16* __restrict__ Qh, const _Float16* __restrict__ Kh,
    float* __restrict__ attn) {
  int bp = blockIdx.x / MT, mt = blockIdx.x % MT;
  const _Float16* Q = Qh + (bp * Ss + mt * 16) * Dd;
  const _Float16* K = Kh + bp * Ss * Dd;
  float* out = attn + (size_t)bp * Ss * Ss;
  v16h a0 = a_frag(Q, Dd, 0), a1 = a_frag(Q, Dd, 32);
  int l = lane_id(), h = l >> 4, n = l & 15;
  for (int nt = 0; nt <= mt; ++nt) {          // causal: lower tiles only
    const _Float16* Kt = K + nt * 16 * Dd;
    __builtin_prefetch(Kt + 16 * Dd, 0, 1);   // global_prefetch next K tile
    v16h b0 = bT_frag_h(Kt, Dd, 0);           // B[k=d][n=s] = K[s][d], contiguous
    v16h b1 = bT_frag_h(Kt, Dd, 32);
    v8f c = {};
    c = wmma32(a0, b0, c);
    c = wmma32(a1, b1, c);
#pragma unroll
    for (int f = 0; f < 8; ++f) {
      int r = mt * 16 + f + 8 * h;
      out[r * Ss + nt * 16 + n] = c[f];
    }
  }
}

// ---------------- Kernel 4: in-place masked softmax (wave32 row ownership) ----
__global__ void smc_softmax_kernel(float* __restrict__ attn_io,
                                   _Float16* __restrict__ attn_h) {
  int wave = threadIdx.x >> 5, lane = threadIdx.x & 31;
  int row = blockIdx.x * 4 + wave;            // global token row 0..12799
  int s = row % Ss;                           // causal horizon
  float* p = attn_io + (size_t)row * Ss;
  float mx = -1e30f;
#pragma unroll
  for (int i = 0; i < 5; ++i) {
    int c = i * 32 + lane;
    if (c <= s) mx = fmaxf(mx, p[c]);
  }
#pragma unroll
  for (int off = 16; off; off >>= 1) mx = fmaxf(mx, __shfl_xor(mx, off, 32));
  float vals[5], sum = 0.0f;
#pragma unroll
  for (int i = 0; i < 5; ++i) {
    int c = i * 32 + lane;
    float v = (c <= s) ? __expf(p[c] - mx) : 0.0f;  // future -> exact 0 (matches -1e9)
    vals[i] = v;
    sum += v;
  }
#pragma unroll
  for (int off = 16; off; off >>= 1) sum += __shfl_xor(sum, off, 32);
  float inv = 1.0f / sum;
#pragma unroll
  for (int i = 0; i < 5; ++i) {
    int c = i * 32 + lane;
    float v = vals[i] * inv;
    p[c] = v;
    attn_h[(size_t)row * Ss + c] = (_Float16)v;
  }
}

// ---------------- Kernel 5: z = attn@V, o = z@Wo + bo, LN1(o + x) ----------------
__global__ void __launch_bounds__(32) smc_zout_kernel(
    const _Float16* __restrict__ attn_h, const _Float16* __restrict__ VhT,
    const _Float16* __restrict__ Wop, const float* __restrict__ bo,
    const float* __restrict__ x32,
    const float* __restrict__ g1, const float* __restrict__ be1,
    float* __restrict__ out32, _Float16* __restrict__ out16) {
  __shared__ _Float16 zh[16 * Dd];
  __shared__ float ts[16 * Dd];
  int bp = blockIdx.x / MT, mt = blockIdx.x % MT;
  int l = lane_id(), h = l >> 4, n = l & 15;
  const _Float16* At = attn_h + (size_t)(bp * Ss + mt * 16) * Ss;
  const _Float16* VT = VhT + bp * Dd * Ss;     // [d][s]
  v16h af[5];
#pragma unroll
  for (int kc = 0; kc < 5; ++kc) af[kc] = a_frag(At, Ss, kc * 32);  // K = 160
#pragma unroll
  for (int nt = 0; nt < 4; ++nt) {
    v8f c = {};
#pragma unroll
    for (int kc = 0; kc < 5; ++kc)             // B[k=s][n=d] = VT[d*Ss + s]
      c = wmma32(af[kc], bT_frag_h(VT + nt * 16 * Ss, Ss, kc * 32), c);
#pragma unroll
    for (int f = 0; f < 8; ++f) zh[(f + 8 * h) * Dd + nt * 16 + n] = (_Float16)c[f];
  }
  __syncthreads();
  v16h za0 = a_frag(zh, Dd, 0), za1 = a_frag(zh, Dd, 32);
  const float* xr = x32 + (bp * Ss + mt * 16) * Dd;
#pragma unroll
  for (int nt = 0; nt < 4; ++nt) {
    v8f c = {};
    c = wmma32(za0, b_frag_packed(Wop, 0, nt, 4), c);
    c = wmma32(za1, b_frag_packed(Wop, 1, nt, 4), c);
    int ng = nt * 16 + n;
    float bb = bo[ng];
#pragma unroll
    for (int f = 0; f < 8; ++f) {
      int m = f + 8 * h;
      ts[m * Dd + ng] = c[f] + bb + xr[m * Dd + ng];  // residual
    }
  }
  __syncthreads();
  if (l < 16) {                                        // LayerNorm row-per-lane
    const float* row = ts + l * Dd;
    float mu = 0.0f;
    for (int d = 0; d < Dd; ++d) mu += row[d];
    mu *= (1.0f / Dd);
    float var = 0.0f;
    for (int d = 0; d < Dd; ++d) { float dd = row[d] - mu; var += dd * dd; }
    var *= (1.0f / Dd);
    float inv = rsqrtf(var + 1e-6f);
    int gr = (bp * Ss + mt * 16 + l) * Dd;
    for (int d = 0; d < Dd; ++d) {
      float y = g1[d] * (row[d] - mu) * inv + be1[d];
      out32[gr + d] = y;
      out16[gr + d] = (_Float16)y;
    }
  }
}

// ------- Kernel 6: FFN 64->256->64, LN2(residual), R out, pred = r@Wf + bf -------
__global__ void __launch_bounds__(32) smc_ffn_kernel(
    const _Float16* __restrict__ out16, const float* __restrict__ out32,
    const _Float16* __restrict__ W1p, const float* __restrict__ b1,
    const _Float16* __restrict__ W2p, const float* __restrict__ b2,
    const float* __restrict__ g2, const float* __restrict__ be2,
    const float* __restrict__ Wf, const float* __restrict__ bf,
    float* __restrict__ Rout, float* __restrict__ pred, float* __restrict__ pred2) {
  __shared__ _Float16 r1h[16 * DFFf];
  __shared__ float ts[16 * Dd];
  int bp = blockIdx.x / MT, mt = blockIdx.x % MT;
  int l = lane_id(), h = l >> 4, n = l & 15;
  const _Float16* A = out16 + (bp * Ss + mt * 16) * Dd;
  v16h a0 = a_frag(A, Dd, 0), a1 = a_frag(A, Dd, 32);
#pragma unroll
  for (int nt = 0; nt < 16; ++nt) {                    // N = 256
    v8f c = {};
    c = wmma32(a0, b_frag_packed(W1p, 0, nt, 16), c);
    c = wmma32(a1, b_frag_packed(W1p, 1, nt, 16), c);
    int ng = nt * 16 + n;
    float bv = b1[ng];
#pragma unroll
    for (int f = 0; f < 8; ++f)
      r1h[(f + 8 * h) * DFFf + ng] = (_Float16)fmaxf(c[f] + bv, 0.0f);  // relu
  }
  __syncthreads();
  const float* orow = out32 + (bp * Ss + mt * 16) * Dd;
#pragma unroll
  for (int nt = 0; nt < 4; ++nt) {                     // K = 256 back to 64
    v8f c = {};
#pragma unroll
    for (int kc = 0; kc < 8; ++kc)
      c = wmma32(a_frag(r1h, DFFf, kc * 32), b_frag_packed(W2p, kc, nt, 4), c);
    int ng = nt * 16 + n;
    float bv = b2[ng];
#pragma unroll
    for (int f = 0; f < 8; ++f) {
      int m = f + 8 * h;
      ts[m * Dd + ng] = c[f] + bv + orow[m * Dd + ng];  // residual
    }
  }
  __syncthreads();
  if (l < 16) {
    const float* row = ts + l * Dd;
    float mu = 0.0f;
    for (int d = 0; d < Dd; ++d) mu += row[d];
    mu *= (1.0f / Dd);
    float var = 0.0f;
    for (int d = 0; d < Dd; ++d) { float dd = row[d] - mu; var += dd * dd; }
    var *= (1.0f / Dd);
    float inv = rsqrtf(var + 1e-6f);
    int grow = bp * Ss + mt * 16 + l;
    float p = bf[0];
    for (int d = 0; d < Dd; ++d) {
      float y = g2[d] * (row[d] - mu) * inv + be2[d];
      Rout[grow * Dd + d] = y;
      p += y * Wf[d];                                  // final head D->1
    }
    pred[grow]  = p;
    pred2[grow] = p;                                   // identity resampling
  }
}

// -------------------------------- Host launcher --------------------------------
extern "C" void kernel_launch(void* const* d_in, const int* in_sizes, int n_in,
                              void* d_out, int out_size, void* d_ws, size_t ws_size,
                              hipStream_t stream) {
  const float* inputs = (const float*)d_in[0];
  // d_in[1] = targets (unused by forward outputs)
  const float* Wp  = (const float*)d_in[2];  const float* bpj = (const float*)d_in[3];
  const float* Wq  = (const float*)d_in[4];  const float* bq  = (const float*)d_in[5];
  const float* Wk  = (const float*)d_in[6];  const float* bk  = (const float*)d_in[7];
  const float* Wv  = (const float*)d_in[8];  const float* bv  = (const float*)d_in[9];
  const float* Wo  = (const float*)d_in[10]; const float* bo  = (const float*)d_in[11];
  const float* g1  = (const float*)d_in[12]; const float* be1 = (const float*)d_in[13];
  const float* W1  = (const float*)d_in[14]; const float* b1  = (const float*)d_in[15];
  const float* W2  = (const float*)d_in[16]; const float* b2  = (const float*)d_in[17];
  const float* g2  = (const float*)d_in[18]; const float* be2 = (const float*)d_in[19];
  const float* Wf  = (const float*)d_in[20]; const float* bf  = (const float*)d_in[21];

  // workspace carve-up (all offsets >= 32B aligned)
  char* ws = (char*)d_ws;
  size_t o = 0;
  float*    x32    = (float*)(ws + o);    o += (size_t)ROWS * Dd * 4;
  _Float16* x16    = (_Float16*)(ws + o); o += (size_t)ROWS * Dd * 2;
  _Float16* Qh     = (_Float16*)(ws + o); o += (size_t)ROWS * Dd * 2;
  _Float16* Kh     = (_Float16*)(ws + o); o += (size_t)ROWS * Dd * 2;
  _Float16* VhT    = (_Float16*)(ws + o); o += (size_t)ROWS * Dd * 2;  // [bp][d][s]
  _Float16* attn_h = (_Float16*)(ws + o); o += (size_t)ROWS * Ss * 2;
  float*    out32  = (float*)(ws + o);    o += (size_t)ROWS * Dd * 4;
  _Float16* out16  = (_Float16*)(ws + o); o += (size_t)ROWS * Dd * 2;
  _Float16* Wqp    = (_Float16*)(ws + o); o += (size_t)Dd * Dd * 2;
  _Float16* Wkp    = (_Float16*)(ws + o); o += (size_t)Dd * Dd * 2;
  _Float16* Wvp    = (_Float16*)(ws + o); o += (size_t)Dd * Dd * 2;
  _Float16* Wop    = (_Float16*)(ws + o); o += (size_t)Dd * Dd * 2;
  _Float16* W1p    = (_Float16*)(ws + o); o += (size_t)Dd * DFFf * 2;
  _Float16* W2p    = (_Float16*)(ws + o); o += (size_t)DFFf * Dd * 2;

  // output carve-up: pred, pred_resampl, K, V, R, attn_weights (flat, f32)
  float* out   = (float*)d_out;
  float* pred  = out;                       // 12800
  float* pred2 = out + ROWS;                // 12800
  float* Kout  = out + 2 * ROWS;            // 819200
  float* Vout  = Kout + (size_t)ROWS * Dd;  // 819200
  float* Rout  = Vout + (size_t)ROWS * Dd;  // 819200
  float* attn  = Rout + (size_t)ROWS * Dd;  // 2048000

  // pack weights to f16 fragment-major (tiny, one-shot)
  smc_packw_kernel<<<(Dd * Dd + 255) / 256, 256, 0, stream>>>(Wq, Wqp, Dd, Dd);
  smc_packw_kernel<<<(Dd * Dd + 255) / 256, 256, 0, stream>>>(Wk, Wkp, Dd, Dd);
  smc_packw_kernel<<<(Dd * Dd + 255) / 256, 256, 0, stream>>>(Wv, Wvp, Dd, Dd);
  smc_packw_kernel<<<(Dd * Dd + 255) / 256, 256, 0, stream>>>(Wo, Wop, Dd, Dd);
  smc_packw_kernel<<<(Dd * DFFf + 255) / 256, 256, 0, stream>>>(W1, W1p, Dd, DFFf);
  smc_packw_kernel<<<(DFFf * Dd + 255) / 256, 256, 0, stream>>>(W2, W2p, DFFf, Dd);

  smc_proj_kernel<<<(ROWS * Dd) / 256, 256, 0, stream>>>(inputs, Wp, bpj, x32, x16);
  smc_qkv_kernel<<<ROWS / 16, 32, 0, stream>>>(x16, Wqp, bq, Wkp, bk, Wvp, bv,
                                               Qh, Kh, VhT, Kout, Vout);
  smc_scores_kernel<<<NBP * MT, 32, 0, stream>>>(Qh, Kh, attn);
  smc_softmax_kernel<<<ROWS / 4, 128, 0, stream>>>(attn, attn_h);
  smc_zout_kernel<<<NBP * MT, 32, 0, stream>>>(attn_h, VhT, Wop, bo, x32, g1, be1,
                                               out32, out16);
  smc_ffn_kernel<<<NBP * MT, 32, 0, stream>>>(out16, out32, W1p, b1, W2p, b2,
                                              g2, be2, Wf, bf, Rout, pred, pred2);
}
